// HungarianMatcher_91250875171593
// MI455X (gfx1250) — matrix operand order repaired and checked by reading, not compiled, not run on hardware
//
#include <hip/hip_runtime.h>
#include <hip/hip_bf16.h>

// Sizes from the reference
#define BB 4
#define QQ 300
#define CC_DIM 81
#define PT 50176
#define GG 100
#define NP 12544

#define QTILES 19           // ceil(300/16)
#define GPAD 128            // 8 waves * 16
#define KCHUNK 256
#define BLOCK 256

typedef _Float16 v8h  __attribute__((ext_vector_type(8)));
typedef _Float16 v16h __attribute__((ext_vector_type(16)));
typedef float    v8f  __attribute__((ext_vector_type(8)));

__device__ __forceinline__ float sigmoidf_fast(float v) {
    // single v_rcp_f32 instead of the div_scale/fixup sequence
    return __builtin_amdgcn_rcpf(1.0f + __expf(-v));
}

// ---------------------------------------------------------------------------
// 1) cost_class[b,q,g] = -softmax(pred_logits[b,q,:])[clip(gt_labels[b,g])]
// grid = B*Q blocks, 128 threads
// ---------------------------------------------------------------------------
__global__ __launch_bounds__(128) void class_cost_kernel(
    const float* __restrict__ logits, const int* __restrict__ labels,
    float* __restrict__ cc)
{
    __shared__ float se[CC_DIM];
    __shared__ float red[128];
    const int bq = blockIdx.x;           // b*300 + q
    const int b  = bq / QQ;
    const int t  = threadIdx.x;

    float l = (t < CC_DIM) ? logits[(size_t)bq * CC_DIM + t] : -1e30f;
    red[t] = l;
    __syncthreads();
    for (int s = 64; s > 0; s >>= 1) {
        if (t < s) red[t] = fmaxf(red[t], red[t + s]);
        __syncthreads();
    }
    const float mx = red[0];
    __syncthreads();

    float e = (t < CC_DIM) ? __expf(l - mx) : 0.0f;
    if (t < CC_DIM) se[t] = e;
    red[t] = e;
    __syncthreads();
    for (int s = 64; s > 0; s >>= 1) {
        if (t < s) red[t] += red[t + s];
        __syncthreads();
    }
    const float inv_sum = __builtin_amdgcn_rcpf(red[0]);

    if (t < GG) {
        int lab = labels[b * GG + t];
        lab = lab < 0 ? 0 : (lab > CC_DIM - 1 ? CC_DIM - 1 : lab);
        cc[(size_t)bq * GG + t] = -se[lab] * inv_sum;
    }
}

// ---------------------------------------------------------------------------
// 2) per-(b,q): xsum = sum_p sigmoid(x), spmean = mean_p softplus(sigmoid(x))
// grid = B*Q blocks, 256 threads
// ---------------------------------------------------------------------------
__global__ __launch_bounds__(256) void row_stats_kernel(
    const float* __restrict__ pred_masks, const int* __restrict__ point_idx,
    float* __restrict__ spmean, float* __restrict__ xsum)
{
    const int bq = blockIdx.x;
    const int b  = bq / QQ;
    const int t  = threadIdx.x;
    const float* __restrict__ row = pred_masks + (size_t)bq * PT;
    const int* __restrict__ idx   = point_idx + b * NP;

    float sx = 0.0f, sp = 0.0f;
    for (int p = t; p < NP; p += 256) {
        float x = sigmoidf_fast(row[idx[p]]);
        sx += x;
        sp += __logf(1.0f + __expf(x));   // softplus(x), x in (0,1): safe
    }
    __shared__ float r1[256], r2[256];
    r1[t] = sx; r2[t] = sp;
    __syncthreads();
    for (int s = 128; s > 0; s >>= 1) {
        if (t < s) { r1[t] += r1[t + s]; r2[t] += r2[t + s]; }
        __syncthreads();
    }
    if (t == 0) { xsum[bq] = r1[0]; spmean[bq] = r2[0] * (1.0f / NP); }
}

// ---------------------------------------------------------------------------
// 3) per-(b,g): gsum = sum_p gt_mask gathered
// grid = B*G blocks, 256 threads
// ---------------------------------------------------------------------------
__global__ __launch_bounds__(256) void gsum_kernel(
    const int* __restrict__ gt_masks, const int* __restrict__ point_idx,
    float* __restrict__ gsum)
{
    const int bg = blockIdx.x;
    const int b  = bg / GG;
    const int t  = threadIdx.x;
    const int* __restrict__ row = gt_masks + (size_t)bg * PT;
    const int* __restrict__ idx = point_idx + b * NP;

    float s = 0.0f;
    for (int p = t; p < NP; p += 256) s += (float)row[idx[p]];
    __shared__ float r[256];
    r[t] = s;
    __syncthreads();
    for (int ss = 128; ss > 0; ss >>= 1) {
        if (t < ss) r[t] += r[t + ss];
        __syncthreads();
    }
    if (t == 0) gsum[bg] = r[0];
}

// ---------------------------------------------------------------------------
// 4) WMMA GEMM xg[b,q,g] = sum_p sigmoid(pm)[q,p] * gm[g,p], fused epilogue.
// grid = B*19 workgroups, 256 threads (8 waves; wave w owns g-tile w, tile 7
// is zero padding). Each thread owns one K-column per chunk: its gather index
// lives in a register, so staging is pure strided global gathers + ds stores.
// ---------------------------------------------------------------------------
__global__ __launch_bounds__(BLOCK) void wmma_cost_kernel(
    const float* __restrict__ pred_masks, const int* __restrict__ gt_masks,
    const int* __restrict__ point_idx,
    const float* __restrict__ cc, const float* __restrict__ spmean,
    const float* __restrict__ xsum, const float* __restrict__ gsum,
    float* __restrict__ out)
{
    __shared__ _Float16 lds_a[16 * KCHUNK];     // A: 16 q-rows x 256 k   (8 KB)
    __shared__ _Float16 lds_b[GPAD * KCHUNK];   // B: 128 g-rows x 256 k  (64 KB)

    const int wg   = blockIdx.x;                // 0..75
    const int b    = wg / QTILES;
    const int q0   = (wg % QTILES) * 16;
    const int t    = threadIdx.x;               // == my K-column j in [0,256)
    const int wave = t >> 5;
    const int lane = t & 31;

    const float* __restrict__ pm_base = pred_masks + (size_t)b * QQ * PT;
    const int*   __restrict__ gm_base = gt_masks  + (size_t)b * GG * PT;
    const int*   __restrict__ idx     = point_idx + b * NP;

    v8f acc = {};

    // Fragment addressing per CDNA5 ISA layouts (wave32):
    //  A 16x32 f16: lanes 0-15 (m=lane):  h0..7 = K0..7,  h8..15 = K16..23
    //               lanes 16-31 (m=lane-16): K8..15 / K24..31
    //  B 32x16 f16: lanes 0-15 col n=lane K0..7, K16..23; lanes 16-31 col n=lane-16 K8..15, K24..31
    const int am   = lane & 15;
    const int aoff = (lane >> 4) * 8;
    const int boff = (lane >> 4) * 8;
    const int brow = wave * 16 + (lane & 15);

    for (int p0 = 0; p0 < NP; p0 += KCHUNK) {
        // coalesced; one register-resident gather index per thread per chunk
        const int myidx = idx[p0 + t];

        __syncthreads();                        // prev chunk's LDS reads done

        // Stage A tile: 16 rows of my column (gather + sigmoid -> f16)
        #pragma unroll 4
        for (int r = 0; r < 16; ++r) {
            const int q = q0 + r;
            float v = 0.0f;
            if (q < QQ) v = sigmoidf_fast(pm_base[(size_t)q * PT + myidx]);
            lds_a[r * KCHUNK + t] = (_Float16)v;
        }
        // Stage B tile: 128 rows of my column (gt mask 0/1 exact in f16)
        #pragma unroll 8
        for (int gg = 0; gg < GPAD; ++gg) {
            float v = 0.0f;
            if (gg < GG) v = (float)gm_base[(size_t)gg * PT + myidx];
            lds_b[gg * KCHUNK + t] = (_Float16)v;
        }
        __syncthreads();

        // 8 WMMAs cover K=256
        #pragma unroll
        for (int kk = 0; kk < 8; ++kk) {
            const int kb = kk * 32;
            v8h alo = *(const v8h*)&lds_a[am * KCHUNK + kb + aoff];
            v8h ahi = *(const v8h*)&lds_a[am * KCHUNK + kb + 16 + aoff];
            const v8h* pb = (const v8h*)&lds_b[brow * KCHUNK + kb + boff];
            v8h blo = pb[0];
            v8h bhi = pb[2];
            v16h av, bv;
            #pragma unroll
            for (int i = 0; i < 8; ++i) {
                av[i] = alo[i]; av[i + 8] = ahi[i];
                bv[i] = blo[i]; bv[i + 8] = bhi[i];
            }
            acc = __builtin_amdgcn_wmma_f32_16x16x32_f16(
                false, av, false, bv, (short)0, acc, false, false);
        }
    }

    // Epilogue: D layout lane<16 -> rows 0..7, lane>=16 -> rows 8..15; col=lane&15
    const int g = wave * 16 + (lane & 15);
    #pragma unroll
    for (int r = 0; r < 8; ++r) {
        const int m = (lane >> 4) * 8 + r;
        const int q = q0 + m;
        if (q < QQ && g < GG) {
            const int bq = b * QQ + q;
            const float xg = acc[r];
            const float cost_mask = spmean[bq] - xg * (1.0f / NP);
            const float denom = xsum[bq] + gsum[b * GG + g];
            const float cost_dice = 1.0f - (2.0f * xg + 1.0f) / (denom + 1.0f);
            const float cost_cls = cc[(size_t)bq * GG + g];
            out[(size_t)bq * GG + g] =
                2.0f * cost_cls + 5.0f * cost_mask + 5.0f * cost_dice;
        }
    }
}

// ---------------------------------------------------------------------------
extern "C" void kernel_launch(void* const* d_in, const int* in_sizes, int n_in,
                              void* d_out, int out_size, void* d_ws, size_t ws_size,
                              hipStream_t stream) {
    const float* pred_logits = (const float*)d_in[0];
    const float* pred_masks  = (const float*)d_in[1];
    const int*   gt_labels   = (const int*)d_in[2];
    const int*   gt_masks    = (const int*)d_in[3];
    const int*   point_idx   = (const int*)d_in[4];
    float* out = (float*)d_out;

    // Workspace partition (~0.5 MB)
    float* cc     = (float*)d_ws;              // B*Q*G = 120000
    float* spmean = cc + BB * QQ * GG;         // B*Q   = 1200
    float* xsum   = spmean + BB * QQ;          // B*Q   = 1200
    float* gsum   = xsum + BB * QQ;            // B*G   = 400

    class_cost_kernel<<<BB * QQ, 128, 0, stream>>>(pred_logits, gt_labels, cc);
    row_stats_kernel<<<BB * QQ, 256, 0, stream>>>(pred_masks, point_idx, spmean, xsum);
    gsum_kernel<<<BB * GG, 256, 0, stream>>>(gt_masks, point_idx, gsum);
    wmma_cost_kernel<<<BB * QTILES, BLOCK, 0, stream>>>(
        pred_masks, gt_masks, point_idx, cc, spmean, xsum, gsum, out);
}
